// JointLoss_26371099198525
// MI455X (gfx1250) — matrix operand order repaired and checked by reading, not compiled, not run on hardware
//
#include <hip/hip_runtime.h>
#include <hip/hip_bf16.h>
#include <math.h>

typedef __attribute__((ext_vector_type(16))) __bf16 v16bf;
typedef __attribute__((ext_vector_type(8)))  float  v8f;

#define NTOT 8192
#define DDIM 256
#define MTILES (NTOT / 16)          // 512
#define KBLK   (DDIM / 32)          // 8
#define MTPB   4                    // M-tiles per block (64 rows)
#define FRAG_ELEMS ((size_t)MTILES * KBLK * 512)   // 2,097,152 bf16 per layout

#define LOG2E10 14.426950408889634f   // 10 / ln(2): exp(10c-10) = exp2(c*L - L)

// ---------------------------------------------------------------------------
// Kernel 1: row norms, cosine-similarity loss, pack normalized rows as bf16
// into WMMA A-fragment and B-fragment layouts (ISA 7.12.2 16-bit layouts).
__global__ __launch_bounds__(DDIM) void pack_kernel(
    const float* __restrict__ wsi, const float* __restrict__ omic,
    __bf16* __restrict__ Aw, __bf16* __restrict__ Bw,
    __bf16* __restrict__ Ao, __bf16* __restrict__ Bo,
    float* __restrict__ acc)
{
    __shared__ float rw[DDIM], ro[DDIM], rx[DDIM];
    __shared__ float bc[2];
    int r = blockIdx.x, t = threadIdx.x;
    float xw = wsi[(size_t)r * DDIM + t];
    float xo = omic[(size_t)r * DDIM + t];
    rw[t] = xw * xw; ro[t] = xo * xo; rx[t] = xw * xo;
    __syncthreads();
    for (int off = DDIM / 2; off > 0; off >>= 1) {
        if (t < off) { rw[t] += rw[t+off]; ro[t] += ro[t+off]; rx[t] += rx[t+off]; }
        __syncthreads();
    }
    if (t == 0) {
        float nw = sqrtf(rw[0]), no = sqrtf(ro[0]);
        float cosv = rx[0] / (fmaxf(nw, 1e-8f) * fmaxf(no, 1e-8f));
        cosv = fminf(fmaxf(cosv, -1.f), 1.f);
        atomicAdd(acc + 0, 1.f - cosv);
        bc[0] = 1.f / fmaxf(nw, 1e-12f);
        bc[1] = 1.f / fmaxf(no, 1e-12f);
    }
    __syncthreads();
    float zw = xw * bc[0], zo = xo * bc[1];

    int mt = r >> 4, m = r & 15, kb = t >> 5, kk = t & 31;
    // A (16x32 bf16): lane<16 holds K=0..7,16..23 ; lane>=16 holds K=8..15,24..31
    int laneA = m + (((kk >> 3) & 1) << 4);
    int iA    = ((kk >> 4) << 3) + (kk & 7);
    size_t idxA = ((size_t)(mt * KBLK + kb)) * 512 + laneA * 16 + iA;
    // B (32x16 bf16): lane<16 holds K=0..7,16..23 ; lane>=16 holds K=8..15,24..31
    int laneB = m + (((kk >> 3) & 1) << 4);
    int iB    = ((kk >> 4) << 3) + (kk & 7);
    size_t idxB = ((size_t)(mt * KBLK + kb)) * 512 + laneB * 16 + iB;

    Aw[idxA] = (__bf16)zw;  Bw[idxB] = (__bf16)zw;
    Ao[idxA] = (__bf16)zo;  Bo[idxB] = (__bf16)zo;
}

// ---------------------------------------------------------------------------
// Kernel 2: one block — bitonic sort times (ascending) in LDS, quantiles,
// risk-group assignment + group histogram, prefix-scan cumsum -> Cox loss.
__global__ __launch_bounds__(1024) void sort_cox_groups_kernel(
    const float* __restrict__ log_risks, const float* __restrict__ times,
    const int* __restrict__ censor, int* __restrict__ groups,
    int* __restrict__ hist, float* __restrict__ acc)
{
    __shared__ float ts[NTOT];
    __shared__ int   si[NTOT];
    __shared__ float red[1024];
    __shared__ float qv[4];
    __shared__ int   ghist[4];
    const int tid = threadIdx.x;
    const int E = NTOT / 1024;   // 8 elements/thread

    if (tid < 4) ghist[tid] = 0;
    for (int e = 0; e < E; ++e) { int i = tid + e * 1024; ts[i] = times[i]; si[i] = i; }
    __syncthreads();

    for (unsigned k = 2; k <= NTOT; k <<= 1) {
        for (unsigned j = k >> 1; j > 0; j >>= 1) {
            for (int e = 0; e < E; ++e) {
                unsigned i = tid + e * 1024;
                unsigned ixj = i ^ j;
                if (ixj > i) {
                    bool up = ((i & k) == 0);
                    float a = ts[i], b = ts[ixj];
                    if ((a > b) == up) {
                        ts[i] = b; ts[ixj] = a;
                        int t2 = si[i]; si[i] = si[ixj]; si[ixj] = t2;
                    }
                }
            }
            __syncthreads();
        }
    }

    if (tid == 0) {   // linear-interp quantiles over ascending array
        for (int g = 1; g <= 4; ++g) {
            float pos = 0.25f * g * (float)(NTOT - 1);
            int lo = (int)floorf(pos); if (lo > NTOT - 1) lo = NTOT - 1;
            int hi = lo + 1; if (hi > NTOT - 1) hi = NTOT - 1;
            float fr = pos - (float)lo;
            qv[g - 1] = ts[lo] + fr * (ts[hi] - ts[lo]);
        }
    }
    __syncthreads();

    for (int e = 0; e < E; ++e) {
        int j = tid + e * 1024;
        float tv = ts[j];
        int g = (tv >= qv[0]) + (tv >= qv[1]) + (tv >= qv[2]) + (tv >= qv[3]);
        g = min(g, 3);
        groups[si[j]] = g;
        atomicAdd(&ghist[g], 1);
    }
    __syncthreads();
    if (tid < 4) hist[tid] = ghist[tid];

    // Cox: descending-time at-risk sums == suffix sums of exp over ascending order
    float loc[8];
    float csum = 0.f;
    const int base = tid * E;
    for (int e = 0; e < E; ++e) {
        float lr = log_risks[si[base + e]];
        lr = fminf(fmaxf(lr, -10.f), 10.f);
        csum += __expf(lr);
        loc[e] = csum;                 // inclusive prefix inside chunk
    }
    red[tid] = csum;
    __syncthreads();
    for (int off = 1; off < 1024; off <<= 1) {
        float v = red[tid] + ((tid >= off) ? red[tid - off] : 0.f);
        __syncthreads();
        red[tid] = v;
        __syncthreads();
    }
    float total = red[1023];
    float cexcl = red[tid] - csum;

    float lsum = 0.f, cnt = 0.f;
    for (int e = 0; e < E; ++e) {
        int j = base + e;
        float lr = fminf(fmaxf(log_risks[si[j]], -10.f), 10.f);
        float Pincl = cexcl + loc[e];
        float ev = loc[e] - ((e == 0) ? 0.f : loc[e - 1]);
        float at_risk = total - Pincl + ev;      // suffix-inclusive sum
        float li = lr - logf(at_risk + 1e-15f);
        if (censor[si[j]] == 1) { lsum -= li; cnt += 1.f; }
    }
    __syncthreads();
    red[tid] = lsum; __syncthreads();
    for (int off = 512; off > 0; off >>= 1) { if (tid < off) red[tid] += red[tid + off]; __syncthreads(); }
    float totloss = red[0];
    __syncthreads();
    red[tid] = cnt; __syncthreads();
    for (int off = 512; off > 0; off >>= 1) { if (tid < off) red[tid] += red[tid + off]; __syncthreads(); }
    if (tid == 0) acc[5] = totloss / fmaxf(red[0], 1.f);
}

// ---------------------------------------------------------------------------
// Kernel 3: fused Z Z^T GEMM (bf16 WMMA) + fixed-shift masked sum-exp epilogue.
// Block = 64 rows (4 M-tiles held in registers per wave), 8 waves split the
// 512 column tiles; each B fragment is reused across the 4 row tiles.
// grid = (128, 2 embeddings), 256 threads.
__global__ __launch_bounds__(256) void gemm_lse_kernel(
    const __bf16* __restrict__ frags, const int* __restrict__ groups,
    const int* __restrict__ hist, float* __restrict__ acc)
{
    __shared__ float lsum[64][128][2];   // 64 KB partial (sp, sa) per row/slot
    const int mt0 = blockIdx.x * MTPB;
    const int emb = blockIdx.y;
    const __bf16* A = frags + (size_t)emb * 2 * FRAG_ELEMS;
    const __bf16* B = A + FRAG_ELEMS;

    const int tid  = threadIdx.x;
    const int wave = tid >> 5;
    const int lane = tid & 31;
    const int hi   = lane >> 4;       // C layout: lanes>=16 hold rows M=8..15
    const int lcol = lane & 15;       // C layout: N = lane & 15

    v16bf a[MTPB][KBLK];
#pragma unroll
    for (int mtl = 0; mtl < MTPB; ++mtl)
#pragma unroll
        for (int kb = 0; kb < KBLK; ++kb)
            a[mtl][kb] = *(const v16bf*)(A + ((size_t)((mt0 + mtl) * KBLK + kb)) * 512 + lane * 16);

    int grow[MTPB][8];
#pragma unroll
    for (int mtl = 0; mtl < MTPB; ++mtl)
#pragma unroll
        for (int v = 0; v < 8; ++v)
            grow[mtl][v] = groups[(mt0 + mtl) * 16 + v + 8 * hi];

    float sp[MTPB][8], sa[MTPB][8];
#pragma unroll
    for (int mtl = 0; mtl < MTPB; ++mtl)
#pragma unroll
        for (int v = 0; v < 8; ++v) { sp[mtl][v] = 0.f; sa[mtl][v] = 0.f; }

    for (int nt = wave; nt < MTILES; nt += 8) {
        v16bf b[KBLK];
#pragma unroll
        for (int kb = 0; kb < KBLK; ++kb)
            b[kb] = *(const v16bf*)(B + ((size_t)(nt * KBLK + kb)) * 512 + lane * 16);

        const int col  = nt * 16 + lcol;
        const int gcol = groups[col];

#pragma unroll
        for (int mtl = 0; mtl < MTPB; ++mtl) {
            v8f c = {0.f, 0.f, 0.f, 0.f, 0.f, 0.f, 0.f, 0.f};
#pragma unroll
            for (int kb = 0; kb < KBLK; ++kb)
                c = __builtin_amdgcn_wmma_f32_16x16x32_bf16(
                        false, a[mtl][kb], false, b[kb], (short)0, c, false, false);
#pragma unroll
            for (int v = 0; v < 8; ++v) {
                const int row = (mt0 + mtl) * 16 + v + 8 * hi;
                // sim in [-10,10] -> fixed shift by -10: exp(10c-10) = exp2(c*L - L)
                float e = exp2f(fmaf(c[v], LOG2E10, -LOG2E10));
                e = (row == col) ? 0.f : e;               // drop diagonal
                sa[mtl][v] += e;                          // all off-diagonal
                sp[mtl][v] += (grow[mtl][v] == gcol) ? e : 0.f;   // positives
            }
        }
    }

    const int slot = wave * 16 + lcol;
#pragma unroll
    for (int mtl = 0; mtl < MTPB; ++mtl)
#pragma unroll
        for (int v = 0; v < 8; ++v) {
            const int ri = mtl * 16 + v + 8 * hi;
            lsum[ri][slot][0] = sp[mtl][v];
            lsum[ri][slot][1] = sa[mtl][v];
        }
    __syncthreads();

    if (tid < 64) {
        float Sp = 0.f, Sa = 0.f;
        for (int s2 = 0; s2 < 128; ++s2) { Sp += lsum[tid][s2][0]; Sa += lsum[tid][s2][1]; }
        const int r = mt0 * 16 + tid;
        const int cg = hist[groups[r]];
        // valid iff row has >=1 positive and >=1 negative partner
        const bool valid = (cg >= 2) && (cg <= NTOT - 1);
        // shifts cancel: lse_all - lse_pos = log(Sa) - log(Sp)
        float loss = valid ? (logf(Sa) - logf(Sp)) : 0.f;
        atomicAdd(acc + 1 + 2 * emb, loss);
        atomicAdd(acc + 2 + 2 * emb, valid ? 1.f : 0.f);
    }
}

// ---------------------------------------------------------------------------
__global__ void finalize_kernel(const float* __restrict__ acc, float* __restrict__ out)
{
    float sim = acc[0] / (float)NTOT;
    float cw  = acc[1] / fmaxf(acc[2], 1.f);
    float co  = acc[3] / fmaxf(acc[4], 1.f);
    out[0] = acc[5] + 1.0f * sim + 0.1f * 0.5f * (cw + co);
}

// ---------------------------------------------------------------------------
extern "C" void kernel_launch(void* const* d_in, const int* in_sizes, int n_in,
                              void* d_out, int out_size, void* d_ws, size_t ws_size,
                              hipStream_t stream)
{
    const float* log_risks = (const float*)d_in[0];
    const float* times     = (const float*)d_in[1];
    const int*   censor    = (const int*)d_in[2];
    const float* wsi       = (const float*)d_in[3];
    const float* omic      = (const float*)d_in[4];

    float*  acc    = (float*)d_ws;                                // 6 floats
    int*    hist   = (int*)((char*)d_ws + 32);                    // 4 ints
    int*    groups = (int*)((char*)d_ws + 256);                   // 8192 ints
    __bf16* frags  = (__bf16*)((char*)d_ws + (1 << 20));          // 4 x 4MB
    __bf16* Aw = frags;
    __bf16* Bw = frags + FRAG_ELEMS;
    __bf16* Ao = frags + 2 * FRAG_ELEMS;
    __bf16* Bo = frags + 3 * FRAG_ELEMS;

    hipMemsetAsync(d_ws, 0, 256, stream);   // zero accumulators (graph-capturable)

    pack_kernel<<<NTOT, DDIM, 0, stream>>>(wsi, omic, Aw, Bw, Ao, Bo, acc);
    sort_cox_groups_kernel<<<1, 1024, 0, stream>>>(log_risks, times, censor, groups, hist, acc);
    dim3 grid(NTOT / (16 * MTPB), 2);
    gemm_lse_kernel<<<grid, 256, 0, stream>>>(frags, groups, hist, acc);
    finalize_kernel<<<1, 1, 0, stream>>>(acc, (float*)d_out);
}
